// DeepBiRNN_6399501271115
// MI455X (gfx1250) — matrix operand order repaired and checked by reading, not compiled, not run on hardware
//
#include <hip/hip_runtime.h>

// ---------------- types ----------------
typedef __attribute__((ext_vector_type(16))) __bf16 v16bf;
typedef __attribute__((ext_vector_type(8)))  __bf16 v8bf;
typedef __attribute__((ext_vector_type(8)))  float  v8f;
typedef unsigned int u32x4 __attribute__((ext_vector_type(4)));
typedef int          i32x4 __attribute__((ext_vector_type(4)));
typedef int          i32x8 __attribute__((ext_vector_type(8)));

#define Bm 32
#define Tm 1024
#define Dm 512
#define Hm 512
#define G4 2048          // 4*H
#define OUTm 1024        // output_size*2
#define NWG 32           // workgroups per direction in the scan

#if defined(__has_builtin)
#if __has_builtin(__builtin_amdgcn_tensor_load_to_lds)
#define HAVE_TDM 1
#endif
#endif

// ---------------- workspace layout (bytes) ----------------
static const size_t XBF_OFF  = 0;                                   // [T,B,D] bf16
static const size_t XBF_SZ   = (size_t)Tm*Bm*Dm*2;                  // 33.5 MB
static const size_t WIT_OFF  = XBF_OFF + XBF_SZ;                    // [dir][l][4H][D] bf16 (transposed)
static const size_t WIT_SZ   = (size_t)2*2*G4*Dm*2;                 // 8.4 MB
static const size_t WHT_OFF  = WIT_OFF + WIT_SZ;                    // [dir][l][4H][H] bf16
static const size_t WHT_SZ   = (size_t)2*2*G4*Hm*2;                 // 8.4 MB
static const size_t WOT_OFF  = WHT_OFF + WHT_SZ;                    // [OUT][2H] bf16
static const size_t WOT_SZ   = (size_t)OUTm*2*Hm*2;                 // 2.1 MB
static const size_t GX_OFF   = WOT_OFF + WOT_SZ;                    // [dir][T*B,4H] f32
static const size_t GX_SZ    = (size_t)2*Tm*Bm*G4*4;                // 537 MB
static const size_t HSEQ_OFF = GX_OFF + GX_SZ;                      // [dir][T*B,H] bf16 (layer0 out)
static const size_t HSEQ_SZ  = (size_t)2*Tm*Bm*Hm*2;                // 67 MB
static const size_t HFB_OFF  = HSEQ_OFF + HSEQ_SZ;                  // [dir][T*B,H] bf16 (layer1 out)
static const size_t HFB_SZ   = (size_t)2*Tm*Bm*Hm*2;                // 67 MB
static const size_t HST_OFF  = HFB_OFF + HFB_SZ;                    // [dir][2][B,H] bf16 double-buffered h
static const size_t HST_SZ   = (size_t)2*2*Bm*Hm*2;
static const size_t CST_OFF  = HST_OFF + HST_SZ;                    // [dir][B,H] f32 cell state
static const size_t CST_SZ   = (size_t)2*Bm*Hm*4;
static const size_t BAR_OFF  = CST_OFF + CST_SZ;                    // [4][T] int barrier counters
static const size_t BAR_SZ   = (size_t)4*Tm*4;

// ---------------- WMMA fragment helpers (wave32, ISA 7.12.2 layouts) ----------------
__device__ __forceinline__ v8f wmma_bf16(v16bf a, v16bf b, v8f c) {
  return __builtin_amdgcn_wmma_f32_16x16x32_bf16(false, a, false, b, (short)0, c, false, false);
}

union FragU { v16bf v; v8bf h[2]; };

// A: 16x32 bf16 tile from row-major [_,lda]; lane L<16: row L, K {0..7,16..23}; L>=16: row L-16, K {8..15,24..31}
__device__ __forceinline__ v16bf load_a_frag(const __bf16* base, int lda, int row0, int kk) {
  int lane = threadIdx.x & 31, lr = lane & 15, hi = (lane >> 4) & 1;
  const __bf16* p = base + (size_t)(row0 + lr) * lda + kk + hi * 8;
  FragU u;
  u.h[0] = *(const v8bf*)p;
  u.h[1] = *(const v8bf*)(p + 16);
  return u.v;
}

// B: 32x16 tile fed from PRE-TRANSPOSED [N,K] source (global or LDS); lane L<16: col L, K {0..7,16..23}; L>=16: col L-16, K {8..15,24..31}
__device__ __forceinline__ v16bf load_b_frag(const __bf16* baseT, int ldb, int col0, int kk) {
  int lane = threadIdx.x & 31, lr = lane & 15, hi = (lane >> 4) & 1;
  const __bf16* p = baseT + (size_t)(col0 + lr) * ldb + kk + hi * 8;
  FragU u;
  u.h[0] = *(const v8bf*)p;
  u.h[1] = *(const v8bf*)(p + 16);
  return u.v;
}

// A frag with fused ReLU (bf16 sign-bit clear)
__device__ __forceinline__ v16bf load_a_frag_relu(const __bf16* base, int lda, int row0, int kk) {
  int lane = threadIdx.x & 31, lr = lane & 15, hi = (lane >> 4) & 1;
  const unsigned short* p = (const unsigned short*)(base + (size_t)(row0 + lr) * lda + kk + hi * 8);
  FragU u;
  unsigned short tmp[16];
#pragma unroll
  for (int e = 0; e < 8; ++e)  { unsigned short s = p[e];      tmp[e]   = (s & 0x8000u) ? 0 : s; }
#pragma unroll
  for (int e = 0; e < 8; ++e)  { unsigned short s = p[16 + e]; tmp[8+e] = (s & 0x8000u) ? 0 : s; }
  unsigned short* d = (unsigned short*)&u;
#pragma unroll
  for (int e = 0; e < 16; ++e) d[e] = tmp[e];
  return u.v;
}

__device__ __forceinline__ float sigm(float x) { return 1.0f / (1.0f + __expf(-x)); }

// ---------------- pack / convert kernels ----------------
__global__ void __launch_bounds__(256) pack_x_k(const float* __restrict__ x, __bf16* __restrict__ xbf) {
  size_t idx = (size_t)blockIdx.x * blockDim.x + threadIdx.x; // dst index over [T,B,D]
  size_t total = (size_t)Tm * Bm * Dm;
  if (idx >= total) return;
  int t  = (int)(idx / (Bm * Dm));
  int r  = (int)(idx % (Bm * Dm));
  int bb = r / Dm, dc = r % Dm;
  xbf[idx] = (__bf16)x[((size_t)bb * Tm + t) * Dm + dc];
}

// dst[n*K + k] = (bf16) src[k*N + n]   (src row-major [K,N])
__global__ void __launch_bounds__(256) transpose_bf16_k(const float* __restrict__ src,
                                                        __bf16* __restrict__ dst, int K, int N) {
  size_t total = (size_t)K * N;
  for (size_t idx = (size_t)blockIdx.x * blockDim.x + threadIdx.x; idx < total;
       idx += (size_t)gridDim.x * blockDim.x) {
    int n = (int)(idx / K), k = (int)(idx % K);
    dst[idx] = (__bf16)src[(size_t)k * N + n];
  }
}

__global__ void __launch_bounds__(256) zero_u32_k(unsigned* __restrict__ p, size_t n) {
  for (size_t i = (size_t)blockIdx.x * blockDim.x + threadIdx.x; i < n;
       i += (size_t)gridDim.x * blockDim.x) p[i] = 0u;
}

// ---------------- gate pre-GEMM: C[M,2048] = A[M,512] @ WiT' + bias ----------------
__global__ void __launch_bounds__(256) gemm_gates_k(const __bf16* __restrict__ A,
                                                    const __bf16* __restrict__ Bt, // [N,K]
                                                    const float* __restrict__ bias,
                                                    float* __restrict__ C,
                                                    int M, int N, int K) {
  int w = threadIdx.x >> 5, wm = w >> 2, wn = w & 3;
  int m0 = blockIdx.y * 64 + wm * 32;
  int n0 = blockIdx.x * 128 + wn * 32;
  int lane = threadIdx.x & 31, lr = lane & 15, hi = (lane >> 4) & 1;
  v8f acc[2][2] = {};
  for (int kk = 0; kk < K; kk += 32) {
    v16bf a0 = load_a_frag(A, K, m0, kk);
    v16bf a1 = load_a_frag(A, K, m0 + 16, kk);
    v16bf b0 = load_b_frag(Bt, K, n0, kk);
    v16bf b1 = load_b_frag(Bt, K, n0 + 16, kk);
    acc[0][0] = wmma_bf16(a0, b0, acc[0][0]);
    acc[0][1] = wmma_bf16(a0, b1, acc[0][1]);
    acc[1][0] = wmma_bf16(a1, b0, acc[1][0]);
    acc[1][1] = wmma_bf16(a1, b1, acc[1][1]);
  }
  float bv0 = bias[n0 + lr], bv1 = bias[n0 + 16 + lr];
#pragma unroll
  for (int i = 0; i < 2; ++i)
#pragma unroll
    for (int e = 0; e < 8; ++e) {
      int row = m0 + i * 16 + e + hi * 8;
      C[(size_t)row * N + n0 + lr]      = acc[i][0][e] + bv0;
      C[(size_t)row * N + n0 + 16 + lr] = acc[i][1][e] + bv1;
    }
}

// ---------------- persistent bidirectional LSTM scan ----------------
// grid = 64 WGs: blockIdx.x/32 = direction, %32 = 16-column slice of H.
// Loop-invariant Wh slice (64 rows x 512 bf16 = 64KB) is staged once into LDS
// via the Tensor Data Mover (one 2D-tile descriptor per gate), then every
// timestep runs [32x512]@[512x64] WMMA with B-fragments from LDS.
__global__ void __launch_bounds__(256) lstm_scan_k(const float* __restrict__ gXbase,
                                                   const __bf16* __restrict__ WhTbase, int layer,
                                                   __bf16* __restrict__ hseqBase,
                                                   __bf16* __restrict__ hstateBase,
                                                   float* __restrict__ cstateBase,
                                                   int* __restrict__ barBase) {
  __shared__ __bf16 ldsWh[64][512];   // 64 KB: rows g*16+lr of this WG's Wh slice
  __shared__ float  smem[4][32][16];  // 8 KB: gate tiles
  const int dir = blockIdx.x >> 5;
  const int wg  = blockIdx.x & 31;
  const int c0  = wg * 16;
  const int w = threadIdx.x >> 5, g = w >> 1, mt = w & 1;
  const int lane = threadIdx.x & 31, lr = lane & 15, hi = (lane >> 4) & 1;

  const float*  gX  = gXbase + (size_t)dir * Tm * Bm * G4;
  const __bf16* WhT = WhTbase + ((size_t)dir * 2 + layer) * G4 * Hm;  // [4H,H]
  __bf16* hseq = hseqBase   + (size_t)dir * Tm * Bm * Hm;
  __bf16* hst  = hstateBase + (size_t)dir * 2 * Bm * Hm;              // double buffer
  float*  cst  = cstateBase + (size_t)dir * Bm * Hm;
  int*    bar  = barBase + dir * Tm;

  const int nrow = g * 512 + c0;  // gate-column base in 4H space

  // ---- preload Wh slice into LDS ----
#ifdef HAVE_TDM
  if (mt == 0) {  // 4 waves, one TDM 2D-tile descriptor per gate
    unsigned ldsOff = (unsigned)(size_t)(&ldsWh[g * 16][0]);
    unsigned long long ga = (unsigned long long)(size_t)(WhT + (size_t)(g * 512 + c0) * Hm);
    u32x4 d0;
    d0[0] = 1u;                                              // count=1 (valid user D#)
    d0[1] = ldsOff;                                          // lds_addr (bytes)
    d0[2] = (unsigned)ga;                                    // global_addr[31:0]
    d0[3] = (unsigned)((ga >> 32) & 0x1FFFFFFu) | (2u << 30); // global_addr[56:32] | type=2
    i32x8 d1;
    d1[0] = (int)(1u << 16);       // data_size=2B; no multicast/pad/iterate
    d1[1] = (int)(512u << 16);     // tensor_dim0 = 512 (elements per row)
    d1[2] = (int)(2048u << 16);    // tensor_dim1 = 2048 rows (OOB guard)
    d1[3] = (int)(512u << 16);     // tile_dim0 = 512
    d1[4] = 16;                    // tile_dim1 = 16 rows
    d1[5] = 512;                   // tensor_dim0_stride = 512 elements
    d1[6] = 0;
    d1[7] = 0;
    i32x4 dz = {0, 0, 0, 0};
#if __clang_major__ >= 23
    i32x8 dz8 = {0, 0, 0, 0, 0, 0, 0, 0};
    __builtin_amdgcn_tensor_load_to_lds(d0, d1, dz, dz, dz8, 0);
#else
    __builtin_amdgcn_tensor_load_to_lds(d0, d1, dz, dz, 0);
#endif
  }
  __builtin_amdgcn_s_wait_tensorcnt(0);
#else
  for (int idx = threadIdx.x; idx < (64 * 512) / 8; idx += 256) {
    int r = idx >> 6, off = (idx & 63) * 8;
    int grow = (r >> 4) * 512 + c0 + (r & 15);
    *(v8bf*)&ldsWh[r][off] = *(const v8bf*)(WhT + (size_t)grow * Hm + off);
  }
#endif
  __syncthreads();

  for (int s = 0; s < Tm; ++s) {
    const int t = dir ? (Tm - 1 - s) : s;
    const __bf16* hread  = hst + (s & 1) * (Bm * Hm);
    __bf16*       hwrite = hst + ((s + 1) & 1) * (Bm * Hm);

    // init accumulator from precomputed x@Wi + b; prefetch next step's slab
    v8f acc;
    const float* gp = gX + ((size_t)t * Bm + mt * 16) * G4 + nrow + lr;
#pragma unroll
    for (int e = 0; e < 8; ++e) acc[e] = gp[(size_t)(e + hi * 8) * G4];
    if (s + 1 < Tm) {
      int tn = dir ? (Tm - 2 - s) : (s + 1);
      __builtin_prefetch(gX + ((size_t)tn * Bm + mt * 16) * G4 + nrow + lr, 0, 1);
    }

    // h_prev @ Wh  (K = 512), B operand from LDS
    for (int kk = 0; kk < Hm; kk += 32) {
      v16bf a = load_a_frag(hread, Hm, mt * 16, kk);
      v16bf b = load_b_frag(&ldsWh[0][0], Hm, g * 16, kk);
      acc = wmma_bf16(a, b, acc);
    }

    // stash gate tile in LDS
#pragma unroll
    for (int e = 0; e < 8; ++e) smem[g][mt * 16 + e + hi * 8][lr] = acc[e];
    __syncthreads();

    // elementwise cell update for this WG's 16 columns (512 elems / 256 thr)
    for (int idx = threadIdx.x; idx < 512; idx += 256) {
      int m = idx >> 4, n = idx & 15;
      float iv = smem[0][m][n], fv = smem[1][m][n];
      float gv = smem[2][m][n], ov = smem[3][m][n];
      float cp = cst[(size_t)m * Hm + c0 + n];
      float cn = sigm(fv) * cp + sigm(iv) * tanhf(gv);
      float hn = sigm(ov) * tanhf(cn);
      cst[(size_t)m * Hm + c0 + n] = cn;
      __bf16 hb = (__bf16)hn;
      hwrite[(size_t)m * Hm + c0 + n] = hb;
      hseq[((size_t)t * Bm + m) * Hm + c0 + n] = hb;
    }
    __syncthreads();

    // device-scope barrier across the 32 WGs of this direction
    if (threadIdx.x == 0) {
      __hip_atomic_fetch_add(bar + s, 1, __ATOMIC_RELEASE, __HIP_MEMORY_SCOPE_AGENT);
      while (__hip_atomic_load(bar + s, __ATOMIC_ACQUIRE, __HIP_MEMORY_SCOPE_AGENT) < NWG)
        __builtin_amdgcn_s_sleep(1);
    }
    __syncthreads();
  }
}

// ---------------- output GEMM: out = relu(concat(hf,hb)) @ Wout + b, with [T,B]->[B,T] remap ----------------
__global__ void __launch_bounds__(256) gemm_out_k(const __bf16* __restrict__ HF,
                                                  const __bf16* __restrict__ HB,
                                                  const __bf16* __restrict__ WoT, // [OUT, 2H]
                                                  const float* __restrict__ bias,
                                                  float* __restrict__ out) {
  const int K = 2 * Hm;  // 1024
  int w = threadIdx.x >> 5, wm = w >> 2, wn = w & 3;
  int m0 = blockIdx.y * 64 + wm * 32;
  int n0 = blockIdx.x * 128 + wn * 32;
  int lane = threadIdx.x & 31, lr = lane & 15, hi = (lane >> 4) & 1;
  v8f acc[2][2] = {};
  for (int kk = 0; kk < K; kk += 32) {
    const __bf16* src = (kk < Hm) ? HF : HB;
    int ko = kk & (Hm - 1);
    v16bf a0 = load_a_frag_relu(src, Hm, m0, ko);
    v16bf a1 = load_a_frag_relu(src, Hm, m0 + 16, ko);
    v16bf b0 = load_b_frag(WoT, K, n0, kk);
    v16bf b1 = load_b_frag(WoT, K, n0 + 16, kk);
    acc[0][0] = wmma_bf16(a0, b0, acc[0][0]);
    acc[0][1] = wmma_bf16(a0, b1, acc[0][1]);
    acc[1][0] = wmma_bf16(a1, b0, acc[1][0]);
    acc[1][1] = wmma_bf16(a1, b1, acc[1][1]);
  }
  float bv0 = bias[n0 + lr], bv1 = bias[n0 + 16 + lr];
#pragma unroll
  for (int i = 0; i < 2; ++i)
#pragma unroll
    for (int e = 0; e < 8; ++e) {
      int row = m0 + i * 16 + e + hi * 8;        // row in [T,B] flattening
      int t = row >> 5, bb = row & 31;
      float* orow = out + ((size_t)bb * Tm + t) * OUTm;
      orow[n0 + lr]      = acc[i][0][e] + bv0;
      orow[n0 + 16 + lr] = acc[i][1][e] + bv1;
    }
}

// ---------------- host ----------------
extern "C" void kernel_launch(void* const* d_in, const int* in_sizes, int n_in,
                              void* d_out, int out_size, void* d_ws, size_t ws_size,
                              hipStream_t stream) {
  (void)in_sizes; (void)n_in; (void)out_size; (void)ws_size;
  const float* x    = (const float*)d_in[0];
  const float* Wi_f = (const float*)d_in[3];
  const float* Wh_f = (const float*)d_in[4];
  const float* b_f  = (const float*)d_in[5];
  const float* Wi_b = (const float*)d_in[6];
  const float* Wh_b = (const float*)d_in[7];
  const float* b_b  = (const float*)d_in[8];
  const float* Wout = (const float*)d_in[9];
  const float* bout = (const float*)d_in[10];

  char* ws = (char*)d_ws;
  __bf16* XBF  = (__bf16*)(ws + XBF_OFF);
  __bf16* WIT  = (__bf16*)(ws + WIT_OFF);
  __bf16* WHT  = (__bf16*)(ws + WHT_OFF);
  __bf16* WOT  = (__bf16*)(ws + WOT_OFF);
  float*  GX   = (float*)(ws + GX_OFF);
  __bf16* HSEQ = (__bf16*)(ws + HSEQ_OFF);
  __bf16* HFB  = (__bf16*)(ws + HFB_OFF);
  __bf16* HST  = (__bf16*)(ws + HST_OFF);
  float*  CST  = (float*)(ws + CST_OFF);
  int*    BAR  = (int*)(ws + BAR_OFF);

  const size_t dirGX   = (size_t)Tm * Bm * G4;
  const size_t dirHSEQ = (size_t)Tm * Bm * Hm;

  // pack x -> bf16 [T,B,D]
  pack_x_k<<<(Tm * Bm * Dm) / 256, 256, 0, stream>>>(x, XBF);

  // transpose+convert weights: WiT/WhT [dir][l][4H][K], WoT [OUT][2H]
  for (int dir = 0; dir < 2; ++dir) {
    const float* Wi = dir ? Wi_b : Wi_f;
    const float* Wh = dir ? Wh_b : Wh_f;
    for (int l = 0; l < 2; ++l) {
      transpose_bf16_k<<<4096, 256, 0, stream>>>(Wi + (size_t)l * Dm * G4,
                                                 WIT + ((size_t)dir * 2 + l) * G4 * Dm, Dm, G4);
      transpose_bf16_k<<<4096, 256, 0, stream>>>(Wh + (size_t)l * Hm * G4,
                                                 WHT + ((size_t)dir * 2 + l) * G4 * Hm, Hm, G4);
    }
  }
  transpose_bf16_k<<<4096, 256, 0, stream>>>(Wout, WOT, 2 * Hm, OUTm);

  dim3 ggrid(G4 / 128, (Tm * Bm) / 64);

  // ---- layer 0: gate pre-GEMMs for both directions, then concurrent bidirectional scan ----
  gemm_gates_k<<<ggrid, 256, 0, stream>>>(XBF, WIT + 0 * (size_t)G4 * Dm, b_f, GX, Tm * Bm, G4, Dm);
  gemm_gates_k<<<ggrid, 256, 0, stream>>>(XBF, WIT + 2 * (size_t)G4 * Dm, b_b, GX + dirGX, Tm * Bm, G4, Dm);
  zero_u32_k<<<64, 256, 0, stream>>>((unsigned*)HST, HST_SZ / 4);
  zero_u32_k<<<64, 256, 0, stream>>>((unsigned*)CST, CST_SZ / 4);
  zero_u32_k<<<8, 256, 0, stream>>>((unsigned*)BAR, 2 * Tm);
  lstm_scan_k<<<64, 256, 0, stream>>>(GX, WHT, 0, HSEQ, HST, CST, BAR);

  // ---- layer 1 ----
  gemm_gates_k<<<ggrid, 256, 0, stream>>>(HSEQ,           WIT + 1 * (size_t)G4 * Hm, b_f + G4, GX, Tm * Bm, G4, Hm);
  gemm_gates_k<<<ggrid, 256, 0, stream>>>(HSEQ + dirHSEQ, WIT + 3 * (size_t)G4 * Hm, b_b + G4, GX + dirGX, Tm * Bm, G4, Hm);
  zero_u32_k<<<64, 256, 0, stream>>>((unsigned*)HST, HST_SZ / 4);
  zero_u32_k<<<64, 256, 0, stream>>>((unsigned*)CST, CST_SZ / 4);
  zero_u32_k<<<8, 256, 0, stream>>>((unsigned*)(BAR + 2 * Tm), 2 * Tm);
  lstm_scan_k<<<64, 256, 0, stream>>>(GX, WHT, 1, HFB, HST, CST, BAR + 2 * Tm);

  // ---- output projection with fused ReLU + [T,B]->[B,T] remap ----
  dim3 ogrid(OUTm / 128, (Tm * Bm) / 64);
  gemm_out_k<<<ogrid, 256, 0, stream>>>(HFB, HFB + dirHSEQ, WOT, bout, (float*)d_out);
}
